// MinkUNet4d_4063039062844
// MI455X (gfx1250) — compile-verified, matrix operand order checked by reading, not run on hardware
//
#include <hip/hip_runtime.h>
#include <hip/hip_bf16.h>

// ---------------- CDNA5 WMMA types ----------------
typedef __attribute__((ext_vector_type(16))) _Float16 v16h;
typedef __attribute__((ext_vector_type(8)))  float    v8f;

union FragH { v16h h; uint4 q[2]; };

// =====================================================================
// Weight pre-swizzle: f32 W[k][ci][co]  ->  f16 B-fragment layout
// dst[(((k*NT+nt)*KT+kt)*16+n)*32+kk]  (kk = K within 32-chunk, n = col in 16-tile)
// so each lane's B fragment is two contiguous 16B loads.
// =====================================================================
__global__ void __launch_bounds__(256) cvt_w_kernel(
    const float* __restrict__ W, int Cin, int Cout, int KT, int NT, int Koffs,
    _Float16* __restrict__ dst)
{
    long long total = (long long)Koffs * NT * KT * 512;
    for (long long idx = (long long)blockIdx.x * blockDim.x + threadIdx.x;
         idx < total; idx += (long long)gridDim.x * blockDim.x) {
        int kk = (int)(idx & 31);
        long long t = idx >> 5;
        int n  = (int)(t & 15); t >>= 4;
        int kt = (int)(t % KT); t /= KT;
        int nt = (int)(t % NT);
        int k  = (int)(t / NT);
        int ci = kt * 32 + kk;
        int co = nt * 16 + n;
        float v = (ci < Cin && co < Cout) ? W[((long long)k * Cin + ci) * Cout + co] : 0.0f;
        dst[idx] = (_Float16)v;
    }
}

// f32 features -> f16 padded (zero pad columns)
__global__ void __launch_bounds__(256) cvt_x_pad(
    const float* __restrict__ x, int C, _Float16* __restrict__ dst, int Kpad, int n)
{
    long long total = (long long)n * Kpad;
    for (long long i = (long long)blockIdx.x * blockDim.x + threadIdx.x;
         i < total; i += (long long)gridDim.x * blockDim.x) {
        int c = (int)(i % Kpad);
        long long r = i / Kpad;
        float v = (c < C) ? x[r * C + c] : 0.0f;
        dst[i] = (_Float16)v;
    }
}

// splits -> (offset k, rowstart) tile descriptors, 16 rows per tile
__global__ void tiles_build(const int* __restrict__ splits, int K,
                            int2* __restrict__ tiles, int* __restrict__ ntiles)
{
    __shared__ int base[132];
    if (threadIdx.x == 0) {
        int acc = 0;
        for (int k = 0; k < K; ++k) {
            base[k] = acc;
            int len = splits[2 * k + 1] - splits[2 * k];
            acc += (len + 15) >> 4;
        }
        base[K] = acc;
        *ntiles = acc;
    }
    __syncthreads();
    for (int k = 0; k < K; ++k) {
        int b0 = base[k], nt = base[k + 1] - b0, s0 = splits[2 * k];
        for (int i = threadIdx.x; i < nt; i += blockDim.x)
            tiles[b0 + i] = make_int2(k, s0 + 16 * i);
    }
}

// =====================================================================
// Gather -> WMMA GEMM -> scatter-add sparse conv tile kernel.
// Block = 128 threads = 4 waves; one 16-row tile; waves stride over
// 16-wide output-channel tiles.
//  * A slab gathered straight into LDS with CDNA5 async-to-LDS VMEM
//    (global_load_async_to_lds_b128, ASYNCcnt), per-lane fragments then
//    read as contiguous ds_load_b128.
//  * B fragments are contiguous global_load_b128 from pre-swizzled
//    weights, warmed with global_prefetch.
//  * Invalid pad rows gather a clamped valid row: D row M depends only
//    on A row M and the scatter is guarded, so no zero-fill is needed.
// IDENT=true: dense rows (1x1 shortcut conv / classifier head).
// =====================================================================
template <bool IDENT>
__global__ void __launch_bounds__(128) spconv_wmma(
    const _Float16* __restrict__ x16, const _Float16* __restrict__ w16,
    const int* __restrict__ inidx, const int* __restrict__ outidx,
    const int2* __restrict__ tiles, const int* __restrict__ ntilesp,
    const int* __restrict__ splits, const int* __restrict__ nrowsp,
    float* __restrict__ out32, int Kpad, int Cout, int NT)
{
    __shared__ _Float16 Ash[16 * 384];
    __shared__ int irow[16];
    __shared__ int orow[16];

    int k, rowstart, m;
    if constexpr (IDENT) {
        int n = *nrowsp;
        rowstart = blockIdx.x * 16;
        if (rowstart >= n) return;
        m = (n - rowstart) < 16 ? (n - rowstart) : 16;
        k = 0;
        if (threadIdx.x < 16) {
            int r = rowstart + threadIdx.x;
            int rc = ((int)threadIdx.x < m) ? r : rowstart;   // clamp pad rows
            irow[threadIdx.x] = rc;
            orow[threadIdx.x] = r;
        }
    } else {
        if ((int)blockIdx.x >= *ntilesp) return;
        int2 t = tiles[blockIdx.x];
        k = t.x; rowstart = t.y;
        int segend = splits[2 * k + 1];
        m = (segend - rowstart) < 16 ? (segend - rowstart) : 16;
        if (threadIdx.x < 16) {
            bool v = (int)threadIdx.x < m;
            int r = rowstart + (v ? (int)threadIdx.x : 0);    // clamp pad rows
            irow[threadIdx.x] = inidx[r];
            orow[threadIdx.x] = v ? outidx[r] : -1;
        }
    }
    __syncthreads();

    const int KT = Kpad >> 5;          // K chunks of 32
    const int chunks = Kpad >> 3;      // 8-half (16B) chunks per row
    // ---- async gather of the 16 x Kpad A slab directly into LDS ----
    for (int idx = threadIdx.x; idx < 16 * chunks; idx += 128) {
        int r = idx / chunks, c = idx % chunks;
        const _Float16* gp = x16 + (long long)irow[r] * Kpad + c * 8;
        unsigned lp = (unsigned)(size_t)(Ash + r * Kpad + c * 8);
        asm volatile("global_load_async_to_lds_b128 %0, %1, off"
                     :: "v"(lp), "v"(gp) : "memory");
    }
    asm volatile("s_wait_asynccnt 0x0" ::: "memory");
    __syncthreads();

    const int lane = threadIdx.x & 31;
    const int wave = threadIdx.x >> 5;
    const int nlo = lane & 15;
    const int hi  = lane >> 4;

    for (int nt = wave; nt < NT; nt += 4) {
        v8f acc = {};
        const _Float16* wb = w16 + (long long)((k * NT + nt) * KT) * 512;
        __builtin_prefetch(wb + (long long)lane * 128, 0, 0);   // warm B panel
        for (int kt = 0; kt < KT; ++kt) {
            FragH a, b;
            const _Float16* arow = Ash + nlo * Kpad + kt * 32;
            a.q[0] = *(const uint4*)(arow + hi * 8);
            a.q[1] = *(const uint4*)(arow + 16 + hi * 8);
            const _Float16* bcol = wb + ((long long)kt * 16 + nlo) * 32;
            b.q[0] = *(const uint4*)(bcol + hi * 8);
            b.q[1] = *(const uint4*)(bcol + 16 + hi * 8);
            acc = __builtin_amdgcn_wmma_f32_16x16x32_f16(
                false, a.h, false, b.h, (short)0, acc, false, false);
        }
        int co = nt * 16 + nlo;
        if (co < Cout) {
            for (int r = 0; r < 8; ++r) {
                int mr = r + 8 * hi;
                if (mr < m) {
                    int orw = orow[mr];
                    if (orw >= 0)
                        atomicAdd(out32 + (long long)orw * Cout + co, acc[r]);
                }
            }
        }
    }
}

// -------- BN column sums (sum, sumsq) --------
__global__ void __launch_bounds__(256) bn_stats(
    const float* __restrict__ x, const int* __restrict__ np, int C,
    float* __restrict__ stats)
{
    int c = threadIdx.x;
    if (c >= C) return;
    int n = *np;
    float s = 0.0f, q = 0.0f;
    for (int r = blockIdx.x; r < n; r += gridDim.x) {
        float v = x[(long long)r * C + c];
        s += v; q += v * v;
    }
    atomicAdd(&stats[c], s);
    atomicAdd(&stats[C + c], q);
}

// -------- fused BN + (BN'd or raw) residual add + ReLU + f16 pad-store --------
__global__ void __launch_bounds__(256) epilogue(
    const float* __restrict__ c32, const float* __restrict__ stats,
    const float* __restrict__ g, const float* __restrict__ b,
    const float* __restrict__ add32, const float* __restrict__ astats,
    const float* __restrict__ ag, const float* __restrict__ ab,
    const _Float16* __restrict__ add16, int KpadAdd,
    _Float16* __restrict__ dst16, int C, int KpadOut,
    const int* __restrict__ np, int relu)
{
    int n = *np;
    float inv = 1.0f / (float)(n > 0 ? n : 1);
    long long total = (long long)n * KpadOut;
    for (long long i = (long long)blockIdx.x * blockDim.x + threadIdx.x;
         i < total; i += (long long)gridDim.x * blockDim.x) {
        int c = (int)(i % KpadOut);
        long long r = i / KpadOut;
        float v = 0.0f;
        if (c < C) {
            float mu  = stats[c] * inv;
            float var = stats[C + c] * inv - mu * mu;
            float sc  = g[c] * rsqrtf(var + 1e-5f);
            v = (c32[r * C + c] - mu) * sc + b[c];
            if (add32) {
                float av = add32[r * C + c];
                if (astats) {
                    float amu  = astats[c] * inv;
                    float avar = astats[C + c] * inv - amu * amu;
                    av = (av - amu) * ag[c] * rsqrtf(avar + 1e-5f) + ab[c];
                }
                v += av;
            } else if (add16) {
                v += (float)add16[r * KpadAdd + c];
            }
            if (relu) v = fmaxf(v, 0.0f);
        }
        dst16[r * KpadOut + c] = (_Float16)v;
    }
}

// -------- channel concat (f16 padded) --------
__global__ void __launch_bounds__(256) concat16(
    const _Float16* __restrict__ a, int Ca, int Kpa,
    const _Float16* __restrict__ b2, int Cb, int Kpb,
    _Float16* __restrict__ d, int Kpo, const int* __restrict__ np)
{
    int n = *np;
    long long total = (long long)n * Kpo;
    for (long long i = (long long)blockIdx.x * blockDim.x + threadIdx.x;
         i < total; i += (long long)gridDim.x * blockDim.x) {
        int c = (int)(i % Kpo);
        long long r = i / Kpo;
        _Float16 v = (_Float16)0.0f;
        if (c < Ca) v = a[r * Kpa + c];
        else if (c < Ca + Cb) v = b2[r * Kpb + (c - Ca)];
        d[i] = v;
    }
}

// =====================================================================
// Host orchestration
// =====================================================================
namespace {
struct BNp  { const float* g; const float* b; };
struct Conv { const float* W; BNp bn; };
struct Res  { const float* W1; BNp b1; const float* W2; BNp b2;
              const float* Wd; BNp bd; };
struct Map  { const int* in; const int* out; const int* spl; const int* np;
              int hits; int K; };
static inline int pad32(int c) { return (c + 31) & ~31; }
static inline int nt16(int c)  { return (c + 15) / 16; }
}

extern "C" void kernel_launch(void* const* d_in, const int* in_sizes, int n_in,
                              void* d_out, int out_size, void* d_ws, size_t ws_size,
                              hipStream_t stream) {
    int ii = 0;
    auto F = [&]() -> const float* { return (const float*)d_in[ii++]; };
    auto getBN  = [&]() { BNp p; p.g = F(); p.b = F(); return p; };
    auto getConv = [&]() { Conv c; c.W = F(); c.bn = getBN(); return c; };
    auto getRes = [&](bool d) {
        Res r; r.W1 = F(); r.b1 = getBN(); r.W2 = F(); r.b2 = getBN();
        if (d) { r.Wd = F(); r.bd = getBN(); } else { r.Wd = nullptr; r.bd = BNp{nullptr, nullptr}; }
        return r;
    };

    const float* feats = F();
    Conv stem = getConv();
    struct Stage { Conv down; Res r1, r2; };
    Stage s1{getConv(), getRes(false), getRes(false)};
    Stage s2{getConv(), getRes(true),  getRes(false)};
    Stage s3{getConv(), getRes(true),  getRes(false)};
    Stage s4{getConv(), getRes(true),  getRes(false)};
    struct UpS { Conv de; Res r1, r2; };
    UpS u1{getConv(), getRes(true), getRes(false)};
    UpS u2{getConv(), getRes(true), getRes(false)};
    UpS u3{getConv(), getRes(true), getRes(false)};
    UpS u4{getConv(), getRes(true), getRes(false)};
    const float* clsW = F();

    auto getMap = [&]() {
        Map m;
        m.hits = in_sizes[ii]; m.in  = (const int*)d_in[ii++];
        m.out  = (const int*)d_in[ii++];
        m.K    = in_sizes[ii] / 2; m.spl = (const int*)d_in[ii++];
        m.np   = (const int*)d_in[ii++];
        return m;
    };
    Map mStem = getMap(), mH0 = getMap(), mH1 = getMap(), mH2 = getMap(),
        mH3 = getMap(), mH4 = getMap(), mD1 = getMap(), mD2 = getMap(),
        mD3 = getMap(), mD4 = getMap(), mU1 = getMap(), mU2 = getMap(),
        mU3 = getMap(), mU4 = getMap();

    const int N0 = in_sizes[0] / 3;
    const long long R = N0;   // row upper bound for every level

    // ---- workspace carve ----
    size_t off = 0;
    auto alloc = [&](size_t bytes) -> void* {
        off = (off + 255) & ~(size_t)255;
        void* p = (char*)d_ws + off;
        off += bytes;
        return p;
    };
    _Float16* S[5];
    for (int i = 0; i < 5; ++i) S[i] = (_Float16*)alloc((size_t)R * 384 * 2);
    _Float16* P0 = (_Float16*)alloc((size_t)R * 32 * 2);
    _Float16* P1 = (_Float16*)alloc((size_t)R * 32 * 2);
    _Float16* P2 = (_Float16*)alloc((size_t)R * 64 * 2);
    _Float16* P3 = (_Float16*)alloc((size_t)R * 128 * 2);
    float* c32   = (float*)alloc((size_t)R * 256 * 4);
    float* sc32  = (float*)alloc((size_t)R * 256 * 4);
    float* stats = (float*)alloc(512 * 4);
    float* astat = (float*)alloc(512 * 4);
    _Float16* w16 = (_Float16*)alloc((size_t)8 << 20);
    int maxHits = 0;
    Map* allMaps[14] = {&mStem,&mH0,&mH1,&mH2,&mH3,&mH4,&mD1,&mD2,&mD3,&mD4,&mU1,&mU2,&mU3,&mU4};
    for (int i = 0; i < 14; ++i) if (allMaps[i]->hits > maxHits) maxHits = allMaps[i]->hits;
    int2* tiles  = (int2*)alloc((size_t)(maxHits / 16 + 256) * sizeof(int2));
    int* ntiles  = (int*)alloc(256);
    if (off > ws_size) return;  // workspace too small: bail out

    // ---- step helpers ----
    auto doConv = [&](const _Float16* x, int Cin, const float* W, int Cout,
                      const Map& m, float* out) {
        int Kp = pad32(Cin), KT = Kp / 32, NT = nt16(Cout);
        hipMemsetAsync(out, 0, (size_t)R * Cout * 4, stream);
        long long wtot = (long long)m.K * NT * KT * 512;
        cvt_w_kernel<<<dim3((unsigned)((wtot + 255) / 256)), 256, 0, stream>>>(
            W, Cin, Cout, KT, NT, m.K, w16);
        tiles_build<<<1, 256, 0, stream>>>(m.spl, m.K, tiles, ntiles);
        int bound = m.hits / 16 + m.K + 1;
        spconv_wmma<false><<<dim3((unsigned)bound), 128, 0, stream>>>(
            x, w16, m.in, m.out, tiles, ntiles, m.spl, nullptr, out, Kp, Cout, NT);
    };
    auto doDense = [&](const _Float16* x, int Cin, const float* W, int Cout,
                       const int* np, float* out, bool zero) {
        int Kp = pad32(Cin), KT = Kp / 32, NT = nt16(Cout);
        if (zero) hipMemsetAsync(out, 0, (size_t)R * Cout * 4, stream);
        long long wtot = (long long)NT * KT * 512;
        cvt_w_kernel<<<dim3((unsigned)((wtot + 255) / 256)), 256, 0, stream>>>(
            W, Cin, Cout, KT, NT, 1, w16);
        spconv_wmma<true><<<dim3((unsigned)((R + 15) / 16)), 128, 0, stream>>>(
            x, w16, nullptr, nullptr, nullptr, nullptr, nullptr, np, out, Kp, Cout, NT);
    };
    auto doStats = [&](const float* x, int C, const int* np, float* st) {
        hipMemsetAsync(st, 0, (size_t)2 * C * 4, stream);
        bn_stats<<<dim3(128), 256, 0, stream>>>(x, np, C, st);
    };
    auto doEpi = [&](const float* x, int C, BNp bn, const int* np,
                     const float* add32, const BNp* abn,
                     const _Float16* add16, int KpadAdd,
                     _Float16* dst, bool relu) {
        doStats(x, C, np, stats);
        const float *as = nullptr, *ag = nullptr, *ab = nullptr;
        if (abn) { doStats(add32, C, np, astat); as = astat; ag = abn->g; ab = abn->b; }
        epilogue<<<dim3(2048), 256, 0, stream>>>(
            x, stats, bn.g, bn.b, add32, as, ag, ab, add16, KpadAdd,
            dst, C, pad32(C), np, relu ? 1 : 0);
    };
    auto doCBR = [&](const _Float16* x, int Cin, const Conv& p, int Cout,
                     const Map& m, const int* np, _Float16* dst) {
        doConv(x, Cin, p.W, Cout, m, c32);
        doEpi(c32, Cout, p.bn, np, nullptr, nullptr, nullptr, 0, dst, true);
    };
    auto doRes = [&](const _Float16* x, int Cin, const Res& p, int Cout,
                     const Map& m, const int* np, _Float16* y1buf, _Float16* dst) {
        doConv(x, Cin, p.W1, Cout, m, c32);
        doEpi(c32, Cout, p.b1, np, nullptr, nullptr, nullptr, 0, y1buf, true);
        doConv(y1buf, Cout, p.W2, Cout, m, c32);
        if (p.Wd) {
            doDense(x, Cin, p.Wd, Cout, np, sc32, true);
            doEpi(c32, Cout, p.b2, np, sc32, &p.bd, nullptr, 0, dst, true);
        } else {
            doEpi(c32, Cout, p.b2, np, nullptr, nullptr, x, pad32(Cin), dst, true);
        }
    };
    auto doCat = [&](const _Float16* a, int Ca, const _Float16* b, int Cb,
                     _Float16* d, const int* np) {
        concat16<<<dim3(2048), 256, 0, stream>>>(
            a, Ca, pad32(Ca), b, Cb, pad32(Cb), d, pad32(Ca + Cb), np);
    };

    // ---- network ----
    cvt_x_pad<<<dim3((unsigned)(((long long)N0 * 32 + 255) / 256)), 256, 0, stream>>>(
        feats, 3, S[0], 32, N0);
    doCBR(S[0], 3, stem, 32, mStem, mStem.np, P0);                    // x0

    doCBR(P0, 32, s1.down, 32, mD1, mD1.np, S[0]);
    doRes(S[0], 32, s1.r1, 32, mH1, mH1.np, S[1], S[2]);
    doRes(S[2], 32, s1.r2, 32, mH1, mH1.np, S[0], P1);                // x1

    doCBR(P1, 32, s2.down, 32, mD2, mD2.np, S[0]);
    doRes(S[0], 32, s2.r1, 64, mH2, mH2.np, S[1], S[2]);
    doRes(S[2], 64, s2.r2, 64, mH2, mH2.np, S[0], P2);                // x2

    doCBR(P2, 64, s3.down, 64, mD3, mD3.np, S[0]);
    doRes(S[0], 64, s3.r1, 128, mH3, mH3.np, S[1], S[2]);
    doRes(S[2], 128, s3.r2, 128, mH3, mH3.np, S[0], P3);              // x3

    doCBR(P3, 128, s4.down, 128, mD4, mD4.np, S[0]);
    doRes(S[0], 128, s4.r1, 256, mH4, mH4.np, S[1], S[2]);
    doRes(S[2], 256, s4.r2, 256, mH4, mH4.np, S[0], S[3]);            // x4 in S3

    doCBR(S[3], 256, u1.de, 256, mU1, mU1.np, S[0]);
    doCat(S[0], 256, P3, 128, S[1], mU1.np);                          // 384
    doRes(S[1], 384, u1.r1, 256, mH3, mH3.np, S[2], S[4]);
    doRes(S[4], 256, u1.r2, 256, mH3, mH3.np, S[0], S[1]);            // y1

    doCBR(S[1], 256, u2.de, 128, mU2, mU2.np, S[0]);
    doCat(S[0], 128, P2, 64, S[3], mU2.np);                           // 192
    doRes(S[3], 192, u2.r1, 128, mH2, mH2.np, S[2], S[4]);
    doRes(S[4], 128, u2.r2, 128, mH2, mH2.np, S[0], S[1]);            // y2

    doCBR(S[1], 128, u3.de, 96, mU3, mU3.np, S[0]);
    doCat(S[0], 96, P1, 32, S[3], mU3.np);                            // 128
    doRes(S[3], 128, u3.r1, 96, mH1, mH1.np, S[2], S[4]);
    doRes(S[4], 96, u3.r2, 96, mH1, mH1.np, S[0], S[1]);              // y3

    doCBR(S[1], 96, u4.de, 96, mU4, mU4.np, S[0]);
    doCat(S[0], 96, P0, 32, S[3], mU4.np);                            // 128
    doRes(S[3], 128, u4.r1, 96, mH0, mH0.np, S[2], S[4]);
    doRes(S[4], 96, u4.r2, 96, mH0, mH0.np, S[0], S[1]);              // y4

    hipMemsetAsync(d_out, 0, (size_t)out_size * 4, stream);
    doDense(S[1], 96, clsW, 20, mStem.np, (float*)d_out, false);      // logits
}
